// s2b_decoder_9199819948138
// MI455X (gfx1250) — matrix operand order repaired and checked by reading, not compiled, not run on hardware
//
#include <hip/hip_runtime.h>
#include <hip/hip_bf16.h>

typedef __attribute__((ext_vector_type(16))) _Float16 v16h;
typedef __attribute__((ext_vector_type(8)))  _Float16 v8h;
typedef __attribute__((ext_vector_type(8)))  float    v8f;

__device__ __forceinline__ v8f wmma_f16(v16h a, v16h b, v8f c) {
    // D = A(16x32 f16) * B(32x16 f16) + C(16x16 f32)
    return __builtin_amdgcn_wmma_f32_16x16x32_f16(
        /*neg_a=*/false, a, /*neg_b=*/false, b,
        /*c_mod=*/(short)0, c, /*reuse_a=*/false, /*reuse_b=*/false);
}

__device__ __forceinline__ v16h concat8(v8h lo, v8h hi) {
    return __builtin_shufflevector(lo, hi, 0, 1, 2, 3, 4, 5, 6, 7,
                                           8, 9, 10, 11, 12, 13, 14, 15);
}

// --- wave32 reductions -------------------------------------------------------
__device__ __forceinline__ float red_sum32(float v) {
    #pragma unroll
    for (int m = 1; m < 32; m <<= 1) v += __shfl_xor(v, m, 32);
    return v;
}
// reduce across the 16 lanes of each half-wave (masks 1,2,4,8 stay in-half)
__device__ __forceinline__ float red_max16(float v) {
    #pragma unroll
    for (int m = 1; m < 16; m <<= 1) v = fmaxf(v, __shfl_xor(v, m, 32));
    return v;
}
__device__ __forceinline__ float red_sum16(float v) {
    #pragma unroll
    for (int m = 1; m < 16; m <<= 1) v += __shfl_xor(v, m, 32);
    return v;
}

// --- CHW -> NC transpose + mask vector --------------------------------------
__global__ void prep_kernel(const float* __restrict__ e, const float* __restrict__ mask,
                            float* __restrict__ X, float* __restrict__ mv,
                            int C, int H, int W, int N) {
    int i = blockIdx.x * blockDim.x + threadIdx.x;
    if (i >= N * C) return;
    int n = i / C, c = i - n * C;
    X[i] = e[c * N + n];
    if (c == 0) {
        int y = n / W, x = n - y * W;
        int ih = (y * 256) / H, iw = (x * 256) / W;
        mv[n] = (mask[ih * 256 + iw] < 1.f) ? 0.f : 1.f;
    }
}

// --- f32 -> f16 staging (weights) -------------------------------------------
__global__ void cvt16_kernel(const float* __restrict__ in, _Float16* __restrict__ out, int n) {
    int i = blockIdx.x * blockDim.x + threadIdx.x;
    if (i < n) out[i] = (_Float16)in[i];
}

// --- LayerNorm: one wave32 per token, f16 output -----------------------------
__global__ void ln_kernel(const float* __restrict__ X, const float* __restrict__ w,
                          const float* __restrict__ b, _Float16* __restrict__ Xn,
                          int N, int C) {
    int lane = threadIdx.x & 31;
    int tok  = blockIdx.x * (blockDim.x >> 5) + (threadIdx.x >> 5);
    if (tok >= N) return;
    const float* x = X + (size_t)tok * C;
    float s = 0.f;
    for (int c = lane; c < C; c += 32) s += x[c];
    float mu = red_sum32(s) / (float)C;
    float v = 0.f;
    for (int c = lane; c < C; c += 32) { float d = x[c] - mu; v += d * d; }
    float inv = rsqrtf(red_sum32(v) / (float)C + 1e-5f);
    _Float16* xo = Xn + (size_t)tok * C;
    for (int c = lane; c < C; c += 32)
        xo[c] = (_Float16)((x[c] - mu) * inv * w[c] + b[c]);
}

// --- WMMA GEMM: Y = A[N x K](f16) * Wm[Cout x K](f16)^T + bias ---------------
// OUT_MODE: 0 = f32 row-major, 1 = f16 row-major, 2 = f16 transposed (Cout x N,
// one contiguous 16B store per lane thanks to the D-layout)
template <int OUT_MODE>
__global__ void gemm_wmma(const _Float16* __restrict__ A, const _Float16* __restrict__ Wm,
                          const float* __restrict__ bias, void* __restrict__ Yv,
                          int N, int Kdim, int Cout) {
    int wave = threadIdx.x >> 5;
    int lane = threadIdx.x & 31;
    int tiles_n = Cout >> 4;
    int total = (N >> 4) * tiles_n;
    int tile = blockIdx.x * (blockDim.x >> 5) + wave;
    if (tile >= total) return;
    int tm = tile / tiles_n, tn = tile - tm * tiles_n;
    int r = lane & 15, half = lane >> 4;
    const _Float16* arow = A  + (size_t)(tm * 16 + r) * Kdim; // A lane holds row r
    const _Float16* wrow = Wm + (size_t)(tn * 16 + r) * Kdim; // B lane holds col r
    v8f acc = {};
    for (int kk = 0; kk < Kdim; kk += 32) {
        // A fragment: K in [half*16,+16)                    -> two b128 loads
        v8h alo = *(const v8h*)(arow + kk + half * 16);
        v8h ahi = *(const v8h*)(arow + kk + half * 16 + 8);
        // B fragment: K in [half*16,+16)                    -> two b128 loads
        v8h blo = *(const v8h*)(wrow + kk + half * 16);
        v8h bhi = *(const v8h*)(wrow + kk + half * 16 + 8);
        acc = wmma_f16(concat8(alo, ahi), concat8(blo, bhi), acc);
    }
    int col = tn * 16 + r;
    float bv = bias[col];
    if (OUT_MODE == 0) {
        float* Y = (float*)Yv;
        #pragma unroll
        for (int j = 0; j < 8; ++j)
            Y[(size_t)(tm * 16 + half * 8 + j) * Cout + col] = acc[j] + bv;
    } else if (OUT_MODE == 1) {
        _Float16* Y = (_Float16*)Yv;
        #pragma unroll
        for (int j = 0; j < 8; ++j)
            Y[(size_t)(tm * 16 + half * 8 + j) * Cout + col] = (_Float16)(acc[j] + bv);
    } else {
        _Float16* Yt = (_Float16*)Yv;   // [Cout x N]
        v8h ov;
        #pragma unroll
        for (int j = 0; j < 8; ++j) ov[j] = (_Float16)(acc[j] + bv);
        *(v8h*)(Yt + (size_t)col * N + tm * 16 + half * 8) = ov;
    }
}

// --- flash-style attention: one wave per (16-row tile, head) -----------------
// streams 32 key/value columns per iteration; V is pre-transposed (Vt[C x N])
template <int D>
__global__ void attn_kernel(const _Float16* __restrict__ Q, const _Float16* __restrict__ Kmat,
                            const _Float16* __restrict__ Vt, const float* __restrict__ mv,
                            _Float16* __restrict__ O, int N, int C) {
    constexpr int KCH = (D + 31) / 32;  // K-chunks for Q*K^T (zero padded)
    constexpr int OCH = (D + 15) / 16;  // 16-col output chunks
    __shared__ _Float16 psh[16 * 32];   // P tile, 16 rows x 32 cols
    int lane = threadIdx.x & 31;
    int r = lane & 15, half = lane >> 4;
    int h   = blockIdx.y;
    int rm0 = blockIdx.x * 16;
    v8h z8 = {};

    // Q fragments (resident): D is a multiple of 8, so each 8-half group is
    // either fully inside the head dim or fully zero padding.
    const _Float16* qrow = Q + (size_t)(rm0 + r) * C + h * D;
    v16h aq[KCH];
    #pragma unroll
    for (int c = 0; c < KCH; ++c) {
        int klo = c * 32 + half * 16;
        int khi = c * 32 + half * 16 + 8;
        v8h lo = (klo < D) ? *(const v8h*)(qrow + klo) : z8;
        v8h hi = (khi < D) ? *(const v8h*)(qrow + khi) : z8;
        aq[c] = concat8(lo, hi);
    }
    float m[8], l[8];
    #pragma unroll
    for (int j = 0; j < 8; ++j) { m[j] = -1e30f; l[j] = 0.f; }
    v8f acc[OCH];
    v8f vzero = {};
    #pragma unroll
    for (int c = 0; c < OCH; ++c) acc[c] = vzero;

    for (int cn0 = 0; cn0 < N; cn0 += 32) {
        // S = Q * K^T for two 16-col tiles (K fragment contiguous per lane)
        const _Float16* krow0 = Kmat + (size_t)(cn0 + r) * C + h * D;
        const _Float16* krow1 = krow0 + (size_t)16 * C;
        v8f s0 = vzero, s1 = vzero;
        #pragma unroll
        for (int c = 0; c < KCH; ++c) {
            int kb0 = c * 32 + half * 16;
            v8h b00 = (kb0     < D) ? *(const v8h*)(krow0 + kb0)     : z8;
            v8h b01 = (kb0 + 8 < D) ? *(const v8h*)(krow0 + kb0 + 8) : z8;
            s0 = wmma_f16(aq[c], concat8(b00, b01), s0);
            v8h b10 = (kb0     < D) ? *(const v8h*)(krow1 + kb0)     : z8;
            v8h b11 = (kb0 + 8 < D) ? *(const v8h*)(krow1 + kb0 + 8) : z8;
            s1 = wmma_f16(aq[c], concat8(b10, b11), s1);
        }
        float mask0 = mv[cn0 + r];       // column masks (lane holds column r of
        float mask1 = mv[cn0 + 16 + r];  //  each 16-col tile)
        float scale[8];
        #pragma unroll
        for (int j = 0; j < 8; ++j) {
            float sj0 = s0[j] * mask0;   // reference zeroes masked scores pre-softmax
            float sj1 = s1[j] * mask1;
            float mnew = fmaxf(m[j], red_max16(fmaxf(sj0, sj1)));
            scale[j] = __expf(m[j] - mnew);
            float p0 = __expf(sj0 - mnew);
            float p1 = __expf(sj1 - mnew);
            l[j] = l[j] * scale[j] + red_sum16(p0 + p1);
            m[j] = mnew;
            // P: D-layout -> row-major 16x32 f16 tile in LDS
            psh[(half * 8 + j) * 32 + r]      = (_Float16)p0;
            psh[(half * 8 + j) * 32 + 16 + r] = (_Float16)p1;
        }
        #pragma unroll
        for (int c = 0; c < OCH; ++c)
            #pragma unroll
            for (int j = 0; j < 8; ++j) acc[c][j] *= scale[j];
        __syncthreads();
        // P A-fragment: full K=32, two aligned ds_load_b128 per lane
        v8h plo = *(const v8h*)(psh + r * 32 + half * 16);
        v8h phi = *(const v8h*)(psh + r * 32 + half * 16 + 8);
        v16h ap = concat8(plo, phi);
        #pragma unroll
        for (int oc = 0; oc < OCH; ++oc) {
            int col = oc * 16 + r;
            // V B-fragment from transposed Vt: K in [half*16,+16) contiguous
            const _Float16* vtcol = Vt + (size_t)(h * D + col) * N + cn0;
            v8h vb0 = (col < D) ? *(const v8h*)(vtcol + half * 16)     : z8;
            v8h vb1 = (col < D) ? *(const v8h*)(vtcol + half * 16 + 8) : z8;
            acc[oc] = wmma_f16(ap, concat8(vb0, vb1), acc[oc]);
        }
        __syncthreads();
    }
    #pragma unroll
    for (int oc = 0; oc < OCH; ++oc) {
        int col = oc * 16 + r;
        if (col < D) {
            #pragma unroll
            for (int j = 0; j < 8; ++j) {
                int row = rm0 + half * 8 + j;
                O[(size_t)row * C + h * D + col] = (_Float16)(acc[oc][j] / l[j]);
            }
        }
    }
}

// --- NC -> CHW ---------------------------------------------------------------
__global__ void untrans_kernel(const float* __restrict__ X, float* __restrict__ out,
                               int N, int C) {
    int i = blockIdx.x * blockDim.x + threadIdx.x;
    if (i >= N * C) return;
    int c = i / N, n = i - c * N;
    out[i] = X[(size_t)n * C + c];
}

extern "C" void kernel_launch(void* const* d_in, const int* in_sizes, int n_in,
                              void* d_out, int out_size, void* d_ws, size_t ws_size,
                              hipStream_t stream) {
    (void)in_sizes; (void)n_in; (void)out_size; (void)ws_size;
    const int dims[4]   = {64, 128, 320, 512};
    const int hs[4]     = {64, 32, 16, 8};
    const int depths[4] = {2, 2, 3, 2};

    const float* mask = (const float*)d_in[4];
    char* wsb = (char*)d_ws;
    // byte offsets (max N*C = 262144 elems; max C*C = 262144 elems)
    float*    X    = (float*)   (wsb + 0);        // 1 MB
    _Float16* XN   = (_Float16*)(wsb + 1048576);  // 512 KB
    _Float16* Qb   = (_Float16*)(wsb + 1572864);  // 512 KB
    _Float16* Kb   = (_Float16*)(wsb + 2097152);  // 512 KB
    _Float16* Vtb  = (_Float16*)(wsb + 2621440);  // 512 KB (transposed V)
    _Float16* At   = (_Float16*)(wsb + 3145728);  // 512 KB
    float*    Mv   = (float*)   (wsb + 3670016);  // 16 KB
    _Float16* Wq16 = (_Float16*)(wsb + 3686400);  // 512 KB
    _Float16* Wk16 = (_Float16*)(wsb + 4210688);  // 512 KB
    _Float16* Wv16 = (_Float16*)(wsb + 4734976);  // 512 KB
    _Float16* Wp16 = (_Float16*)(wsb + 5259264);  // 512 KB
    float* outp = (float*)d_out;

    int pidx = 5;
    size_t ooff = 0;
    for (int s = 0; s < 4; ++s) {
        int C = dims[s], H = hs[s], W = hs[s], N = H * W, D = C / 8;
        const float* e = (const float*)d_in[s];
        prep_kernel<<<(N * C + 255) / 256, 256, 0, stream>>>(e, mask, X, Mv, C, H, W, N);
        for (int blk = 0; blk < depths[s]; ++blk) {
            const float* wq = (const float*)d_in[pidx + 0];
            const float* bq = (const float*)d_in[pidx + 1];
            const float* wk = (const float*)d_in[pidx + 2];
            const float* bk = (const float*)d_in[pidx + 3];
            const float* wv = (const float*)d_in[pidx + 4];
            const float* bv = (const float*)d_in[pidx + 5];
            const float* wp = (const float*)d_in[pidx + 6];
            const float* bp = (const float*)d_in[pidx + 7];
            const float* lw = (const float*)d_in[pidx + 8];
            const float* lb = (const float*)d_in[pidx + 9];
            pidx += 10;

            int cc = C * C, cb = (cc + 255) / 256;
            cvt16_kernel<<<cb, 256, 0, stream>>>(wq, Wq16, cc);
            cvt16_kernel<<<cb, 256, 0, stream>>>(wk, Wk16, cc);
            cvt16_kernel<<<cb, 256, 0, stream>>>(wv, Wv16, cc);
            cvt16_kernel<<<cb, 256, 0, stream>>>(wp, Wp16, cc);

            ln_kernel<<<N / 8, 256, 0, stream>>>(X, lw, lb, XN, N, C);

            int tiles   = (N / 16) * (C / 16);
            int gblocks = (tiles + 7) / 8;    // 8 waves (tiles) per 256-thread block
            gemm_wmma<1><<<gblocks, 256, 0, stream>>>(XN, Wq16, bq, Qb,  N, C, C);
            gemm_wmma<1><<<gblocks, 256, 0, stream>>>(XN, Wk16, bk, Kb,  N, C, C);
            gemm_wmma<2><<<gblocks, 256, 0, stream>>>(XN, Wv16, bv, Vtb, N, C, C);

            dim3 ag(N / 16, 8);
            switch (D) {
                case 8:  attn_kernel<8 ><<<ag, 32, 0, stream>>>(Qb, Kb, Vtb, Mv, At, N, C); break;
                case 16: attn_kernel<16><<<ag, 32, 0, stream>>>(Qb, Kb, Vtb, Mv, At, N, C); break;
                case 40: attn_kernel<40><<<ag, 32, 0, stream>>>(Qb, Kb, Vtb, Mv, At, N, C); break;
                default: attn_kernel<64><<<ag, 32, 0, stream>>>(Qb, Kb, Vtb, Mv, At, N, C); break;
            }
            gemm_wmma<0><<<gblocks, 256, 0, stream>>>(At, Wp16, bp, X, N, C, C);
        }
        untrans_kernel<<<(N * C + 255) / 256, 256, 0, stream>>>(X, outp + ooff, N, C);
        ooff += (size_t)N * C;
    }
}
